// GradientDiffNet_63471026700511
// MI455X (gfx1250) — matrix operand order repaired and checked by reading, not compiled, not run on hardware
//
#include <hip/hip_runtime.h>
#include <hip/hip_bf16.h>

// ---------------------------------------------------------------------------
// GradientDiffNet on MI455X (gfx1250): implicit-GEMM 3x3 convs via
// v_wmma_f32_16x16x32_bf16, wave32, async global->LDS staging.
// ---------------------------------------------------------------------------

#define HH 384
#define WW 384
#define NB 4
#define PLANE ((size_t)NB * HH * WW)   // 589824 pixels
#define DT_STEP 0.1f

typedef __attribute__((ext_vector_type(16))) __bf16 bf16x16;
typedef __attribute__((ext_vector_type(8)))  float  f32x8;

union ABFrag { unsigned u[8]; uint4 q[2]; bf16x16 v; };
union AccFrag { float f[8]; f32x8 v; };

__device__ __forceinline__ unsigned short f2bf(float f) {
    unsigned u = __float_as_uint(f);
    return (unsigned short)((u + 0x7FFFu + ((u >> 16) & 1u)) >> 16);
}

#if __has_builtin(__builtin_amdgcn_global_load_async_to_lds_b32)
#define HAVE_ASYNC_LDS 1
typedef __attribute__((address_space(1))) int* as1_int_ptr;   // global
typedef __attribute__((address_space(3))) int* as3_int_ptr;   // LDS
#else
#define HAVE_ASYNC_LDS 0
#endif

__device__ __forceinline__ void wait_async_lds() {
#if HAVE_ASYNC_LDS
#if __has_builtin(__builtin_amdgcn_s_wait_asynccnt)
    __builtin_amdgcn_s_wait_asynccnt(0);
#else
    asm volatile("s_wait_asynccnt 0x0" ::: "memory");
#endif
#endif
}

// ---------------------------------------------------------------------------
// Weight packing: OIHW f32 -> bf16 pairs in the WMMA B-fragment layout.
// B is K x N (K = Cin*9 padded to chunks of 32, N = 16 per n-tile).
// Layout (16-bit B 32x16, this code's contiguous k order, not the ISA table order): lane%16 = N, lane/16 selects K+16
// half, VGPR v holds K = 2v, 2v+1. Stored as u32[((nt*chunks+c)*32+lane)*8+v].
// ---------------------------------------------------------------------------
__global__ void pack_weights_kernel(const float* __restrict__ W,
                                    unsigned* __restrict__ Bp,
                                    int Cin, int Cout, int chunks, int ntiles)
{
    int t = blockIdx.x * 256 + threadIdx.x;
    int total = ntiles * chunks * 256;
    if (t >= total) return;
    int v    = t & 7;
    int lane = (t >> 3) & 31;
    int c    = (t >> 8) % chunks;
    int nt   = t / (chunks << 8);
    int n    = nt * 16 + (lane & 15);
    int k0   = c * 32 + (lane >> 4) * 16 + 2 * v;
    int Kmax = Cin * 9;
    unsigned packed = 0;
    #pragma unroll
    for (int h = 0; h < 2; ++h) {
        int k = k0 + h;
        float f = 0.0f;
        if (k < Kmax && n < Cout) {
            int cin = k / 9;
            int r   = k - cin * 9;              // ky*3 + kx
            f = W[((size_t)n * Cin + cin) * 9 + r];
        }
        packed |= (unsigned)f2bf(f) << (16 * h);
    }
    Bp[t] = packed;
}

// ---------------------------------------------------------------------------
// Implicit-GEMM conv: one block = (batch b, row y, 32-pixel x segment).
// Stage 0: zero raw LDS tile (borders must read as zero).
// Stage 1: async global->LDS copy of raw bf16 tile, rows y-1..y+1,
//          cols x0-2..x0+33 (4B-aligned dwords; OOB dwords skipped -> zero).
// Stage 2: pack A fragments (LDS->LDS) into the 16-bit A-matrix
//          lane/VGPR layout, contiguous k order as in packed B.
// Stage 3: 8 waves = 2 pixel strips x up-to-4 N-tiles; fully unrolled
//          K-loop of v_wmma_f32_16x16x32_bf16.
// ---------------------------------------------------------------------------
template <int CIN, int CHUNKS, int NTILES, bool RELU, bool BF16OUT>
__global__ __launch_bounds__(256) void conv_wmma_kernel(
    const unsigned short* __restrict__ actIn,   // bf16 [b][CIN][H][W]
    const unsigned* __restrict__ Bpack,         // packed weights
    const float* __restrict__ bias,
    unsigned short* __restrict__ outBf16,       // bf16 out (if BF16OUT)
    float* __restrict__ outF32,                 // f32 out (if !BF16OUT)
    int CoutStore, int CoutStride)
{
    constexpr int TILE_COLS = 36;               // x0-2 .. x0+33
    constexpr int TILE_DW   = TILE_COLS / 2;    // 18 dwords per (cin,row)
    constexpr int TDWORDS   = CIN * 3 * TILE_DW;
    constexpr int KMAX      = CIN * 9;

    __shared__ __align__(16) unsigned tileRaw[TDWORDS];
    __shared__ __align__(16) unsigned Apack[2 * CHUNKS * 32 * 8];

    int bid  = blockIdx.x;
    int xblk = bid % (WW / 32);
    int y    = (bid / (WW / 32)) % HH;
    int b    = bid / ((WW / 32) * HH);
    int x0   = xblk * 32;
    int tid  = threadIdx.x;

    int w    = tid >> 5;
    int lane = tid & 31;
    int s2   = w & 1;          // pixel strip (16 pixels each)
    int nt   = w >> 1;         // N-tile (16 out channels each)

    // Warm this wave's packed-weight tile into all cache levels (WGP scope).
    if (nt < NTILES)
        __builtin_prefetch(&Bpack[(nt * CHUNKS) << 8], 0, 3);

    // ---- stage 0: zero raw tile ------------------------------------------
    #pragma unroll
    for (int d = tid; d < TDWORDS; d += 256) tileRaw[d] = 0;
    __syncthreads();

    // ---- stage 1: async copy of valid dwords -----------------------------
    for (int d = tid; d < TDWORDS; d += 256) {
        int cin = d / (3 * TILE_DW);
        int rem = d - cin * 3 * TILE_DW;
        int ky  = rem / TILE_DW;
        int c2  = rem - ky * TILE_DW;
        int yy  = y + ky - 1;
        int xg  = x0 - 2 + c2 * 2;
        if (yy >= 0 && yy < HH && xg >= 0 && xg + 1 < WW) {
            const unsigned* src = (const unsigned*)
                (actIn + ((size_t)(b * CIN + cin) * HH + yy) * WW + xg);
#if HAVE_ASYNC_LDS
            __builtin_amdgcn_global_load_async_to_lds_b32(
                (as1_int_ptr)src, (as3_int_ptr)&tileRaw[d], 0, 0);
#else
            tileRaw[d] = *src;
#endif
        }
    }
    wait_async_lds();
    __syncthreads();

    // ---- stage 2: pack A fragments (LDS -> LDS) --------------------------
    {
        const unsigned short* tr = (const unsigned short*)tileRaw;
        int v8    = tid & 7;
        int lane8 = tid >> 3;                   // 0..31
        int m8    = lane8 & 15;                 // M within strip
        // A-fragment K offset: VGPR v -> K 2v, 2v+1 (+16 for hi lanes),
        // matching the packed B layout.
        int kbase = (lane8 >> 4) * 16 + v8 * 2;
        #pragma unroll
        for (int s = 0; s < 2; ++s) {
            #pragma unroll
            for (int c = 0; c < CHUNKS; ++c) {
                unsigned packed = 0;
                #pragma unroll
                for (int h = 0; h < 2; ++h) {
                    int k = c * 32 + kbase + h;
                    if (k < KMAX) {
                        int cin = k / 9;
                        int r   = k - cin * 9;
                        int ky  = r / 3;
                        int kx  = r - ky * 3;
                        int col = s * 16 + m8 + kx + 1;   // tile col of x0+s*16+m8+kx-1
                        unsigned short val = tr[(cin * 3 + ky) * TILE_COLS + col];
                        packed |= (unsigned)val << (16 * h);
                    }
                }
                Apack[((s * CHUNKS + c) << 8) + tid] = packed;
            }
        }
    }
    __syncthreads();

    // ---- stage 3: WMMA ----------------------------------------------------
    if (nt >= NTILES) return;

    AccFrag acc;
    #pragma unroll
    for (int i = 0; i < 8; ++i) acc.f[i] = 0.0f;

    #pragma unroll
    for (int c = 0; c < CHUNKS; ++c) {
        ABFrag a, bb;
        const uint4* ap = (const uint4*)&Apack[(((s2 * CHUNKS + c) << 5) + lane) << 3];
        a.q[0] = ap[0];
        a.q[1] = ap[1];
        const uint4* bp = (const uint4*)&Bpack[(((nt * CHUNKS + c) << 5) + lane) << 3];
        bb.q[0] = bp[0];
        bb.q[1] = bp[1];
        acc.v = __builtin_amdgcn_wmma_f32_16x16x32_bf16(
            false, a.v, false, bb.v, (short)0, acc.v, false, false);
    }

    // D layout: lane%16 -> N; VGPR v -> pixel m = (lane/16)*8 + v (contiguous x)
    int n = nt * 16 + (lane & 15);
    if (n >= CoutStore) return;
    float bv = bias[n];
    float r[8];
    #pragma unroll
    for (int i = 0; i < 8; ++i) {
        float t = acc.f[i] + bv;
        r[i] = RELU ? fmaxf(t, 0.0f) : t;
    }
    int xbase = x0 + s2 * 16 + (lane >> 4) * 8;
    size_t idx = ((size_t)(b * CoutStride + n) * HH + y) * WW + xbase;
    if (BF16OUT) {
        uint4 d;
        d.x = f2bf(r[0]) | ((unsigned)f2bf(r[1]) << 16);
        d.y = f2bf(r[2]) | ((unsigned)f2bf(r[3]) << 16);
        d.z = f2bf(r[4]) | ((unsigned)f2bf(r[5]) << 16);
        d.w = f2bf(r[6]) | ((unsigned)f2bf(r[7]) << 16);
        *(uint4*)(outBf16 + idx) = d;               // 16B aligned: idx % 8 == 0
    } else {
        *(float4*)(outF32 + idx)     = make_float4(r[0], r[1], r[2], r[3]);
        *(float4*)(outF32 + idx + 4) = make_float4(r[4], r[5], r[6], r[7]);
    }
}

// ---------------------------------------------------------------------------
// Gradients pass A: g0 = x, g1 = d/dH x, g2 = d/dW x (np.gradient semantics).
// ---------------------------------------------------------------------------
__global__ void grad_a_kernel(const float* __restrict__ X,
                              float* __restrict__ g,
                              unsigned short* __restrict__ act)
{
    size_t p = (size_t)blockIdx.x * 256 + threadIdx.x;
    if (p >= PLANE) return;
    int x = (int)(p % WW);
    int y = (int)((p / WW) % HH);
    int b = (int)(p / ((size_t)HH * WW));

    float c  = X[p];
    float up = (y > 0)      ? X[p - WW] : 0.0f;
    float dn = (y < HH - 1) ? X[p + WW] : 0.0f;
    float dH = (y == 0) ? (dn - c) : (y == HH - 1) ? (c - up) : 0.5f * (dn - up);
    float lf = (x > 0)      ? X[p - 1]  : 0.0f;
    float rt = (x < WW - 1) ? X[p + 1]  : 0.0f;
    float dW = (x == 0) ? (rt - c) : (x == WW - 1) ? (c - lf) : 0.5f * (rt - lf);

    g[p]             = c;
    g[PLANE + p]     = dH;
    g[2 * PLANE + p] = dW;

    size_t q    = p - (size_t)b * HH * WW;
    size_t base = (size_t)b * 6 * HH * WW + q;
    act[base]                        = f2bf(c);
    act[base + (size_t)HH * WW]      = f2bf(dH);
    act[base + 2 * (size_t)HH * WW]  = f2bf(dW);
}

// ---------------------------------------------------------------------------
// Gradients pass B: g3 = dH(g1), g4 = dH(g2), g5 = dW(g2).
// ---------------------------------------------------------------------------
__global__ void grad_b_kernel(float* __restrict__ g,
                              unsigned short* __restrict__ act)
{
    size_t p = (size_t)blockIdx.x * 256 + threadIdx.x;
    if (p >= PLANE) return;
    int x = (int)(p % WW);
    int y = (int)((p / WW) % HH);
    int b = (int)(p / ((size_t)HH * WW));

    const float* g1 = g + PLANE;
    const float* g2 = g + 2 * PLANE;

    float c1  = g1[p];
    float u1  = (y > 0)      ? g1[p - WW] : 0.0f;
    float d1  = (y < HH - 1) ? g1[p + WW] : 0.0f;
    float g3  = (y == 0) ? (d1 - c1) : (y == HH - 1) ? (c1 - u1) : 0.5f * (d1 - u1);

    float c2  = g2[p];
    float u2  = (y > 0)      ? g2[p - WW] : 0.0f;
    float d2  = (y < HH - 1) ? g2[p + WW] : 0.0f;
    float g4  = (y == 0) ? (d2 - c2) : (y == HH - 1) ? (c2 - u2) : 0.5f * (d2 - u2);

    float l2  = (x > 0)      ? g2[p - 1]  : 0.0f;
    float r2  = (x < WW - 1) ? g2[p + 1]  : 0.0f;
    float g5  = (x == 0) ? (r2 - c2) : (x == WW - 1) ? (c2 - l2) : 0.5f * (r2 - l2);

    g[3 * PLANE + p] = g3;
    g[4 * PLANE + p] = g4;
    g[5 * PLANE + p] = g5;

    size_t q    = p - (size_t)b * HH * WW;
    size_t base = (size_t)b * 6 * HH * WW + q;
    act[base + 3 * (size_t)HH * WW] = f2bf(g3);
    act[base + 4 * (size_t)HH * WW] = f2bf(g4);
    act[base + 5 * (size_t)HH * WW] = f2bf(g5);
}

// ---------------------------------------------------------------------------
// Combine: x_out = x_in + DT * sum_j [up*k(j)+down*k(j+1)+left*k(j+2)
//                                     +right*k(j+3) - g*k(j+4)]
// (reference uses overlapping kappa slices j..j+4 -> only channels 0..9).
// ---------------------------------------------------------------------------
__global__ void combine_kernel(const float* __restrict__ Xin,
                               const float* __restrict__ g,
                               const float* __restrict__ kap,
                               float* __restrict__ Xout)
{
    size_t p = (size_t)blockIdx.x * 256 + threadIdx.x;
    if (p >= PLANE) return;
    int x = (int)(p % WW);
    int y = (int)((p / WW) % HH);
    int b = (int)(p / ((size_t)HH * WW));
    size_t q = p - (size_t)b * HH * WW;
    const float* kb = kap + (size_t)b * 10 * HH * WW + q;
    const size_t CH = (size_t)HH * WW;

    float sum = 0.0f;
    #pragma unroll
    for (int j = 0; j < 6; ++j) {
        const float* gp = g + (size_t)j * PLANE;
        float gc = gp[p];
        float up = (y > 0)      ? gp[p - WW] : 0.0f;
        float dn = (y < HH - 1) ? gp[p + WW] : 0.0f;
        float lf = (x > 0)      ? gp[p - 1]  : 0.0f;
        float rt = (x < WW - 1) ? gp[p + 1]  : 0.0f;
        sum += up * kb[(j + 0) * CH] + dn * kb[(j + 1) * CH]
             + lf * kb[(j + 2) * CH] + rt * kb[(j + 3) * CH]
             - gc * kb[(j + 4) * CH];
    }
    Xout[p] = Xin[p] + DT_STEP * sum;
}

// ---------------------------------------------------------------------------
// Host launch
// ---------------------------------------------------------------------------
extern "C" void kernel_launch(void* const* d_in, const int* in_sizes, int n_in,
                              void* d_out, int out_size, void* d_ws, size_t ws_size,
                              hipStream_t stream)
{
    (void)in_sizes; (void)n_in; (void)out_size; (void)ws_size;

    const float* x_in = (const float*)d_in[0];
    const float* Wts[12];
    const float* Bs[12];
    for (int i = 0; i < 12; ++i) {
        Wts[i] = (const float*)d_in[1 + 2 * i];
        Bs[i]  = (const float*)d_in[2 + 2 * i];
    }

    char* w = (char*)d_ws;
    auto alloc = [&](size_t bytes) {
        char* r = w;
        w += (bytes + 255) & ~(size_t)255;
        return r;
    };
    float*          g     = (float*)alloc(6 * PLANE * sizeof(float));
    unsigned short* actA  = (unsigned short*)alloc(64 * PLANE * 2);
    unsigned short* actB  = (unsigned short*)alloc(64 * PLANE * 2);
    float*          kappa = (float*)alloc(10 * PLANE * sizeof(float));
    float*          xping = (float*)alloc(PLANE * sizeof(float));
    float*          xpong = (float*)alloc(PLANE * sizeof(float));

    // conv configs: chans [6,64,64,64,30]; conv3 only needs channels 0..9.
    const int cCin[4]    = {6, 64, 64, 64};
    const int cCout[4]   = {64, 64, 64, 30};
    const int cChunks[4] = {2, 18, 18, 18};
    const int cNt[4]     = {4, 4, 4, 1};

    unsigned* wp[12];
    for (int i = 0; i < 12; ++i) {
        int ci = i % 4;
        wp[i] = (unsigned*)alloc((size_t)cNt[ci] * cChunks[ci] * 256 * 4);
    }
    for (int i = 0; i < 12; ++i) {
        int ci = i % 4;
        int total = cNt[ci] * cChunks[ci] * 256;
        pack_weights_kernel<<<(total + 255) / 256, 256, 0, stream>>>(
            Wts[i], wp[i], cCin[ci], cCout[ci], cChunks[ci], cNt[ci]);
    }

    int pixBlocks  = (int)((PLANE + 255) / 256);
    int convBlocks = NB * HH * (WW / 32);

    const float* xcur = x_in;
    for (int li = 0; li < 3; ++li) {
        grad_a_kernel<<<pixBlocks, 256, 0, stream>>>(xcur, g, actA);
        grad_b_kernel<<<pixBlocks, 256, 0, stream>>>(g, actA);

        int base = li * 4;
        conv_wmma_kernel<6, 2, 4, true, true><<<convBlocks, 256, 0, stream>>>(
            actA, wp[base + 0], Bs[base + 0], actB, (float*)nullptr, 64, 64);
        conv_wmma_kernel<64, 18, 4, true, true><<<convBlocks, 256, 0, stream>>>(
            actB, wp[base + 1], Bs[base + 1], actA, (float*)nullptr, 64, 64);
        conv_wmma_kernel<64, 18, 4, true, true><<<convBlocks, 256, 0, stream>>>(
            actA, wp[base + 2], Bs[base + 2], actB, (float*)nullptr, 64, 64);
        conv_wmma_kernel<64, 18, 1, false, false><<<convBlocks, 256, 0, stream>>>(
            actB, wp[base + 3], Bs[base + 3], (unsigned short*)nullptr, kappa, 10, 10);

        float* xnext = (li == 2) ? (float*)d_out : ((li == 0) ? xping : xpong);
        combine_kernel<<<pixBlocks, 256, 0, stream>>>(xcur, g, kappa, xnext);
        xcur = xnext;
    }
}
